// TFKNRM_Class_24524263260437
// MI455X (gfx1250) — matrix operand order repaired and checked by reading, not compiled, not run on hardware
//
#include <hip/hip_runtime.h>
#include <hip/hip_bf16.h>

typedef __attribute__((ext_vector_type(16))) _Float16 v16h;
typedef __attribute__((ext_vector_type(8)))  _Float16 v8h;
typedef __attribute__((ext_vector_type(8)))  float    v8f;

#define VOCAB  50000
#define D_IN   300
#define D_PAD  320      // padded to 10 K-steps of 32
#define BATCH  256
#define T_LEN  1024
#define Q_LEN  16
#define NKERN  11
#define EMBN_BYTES (VOCAB * D_PAD * 2)    // 32,000,000

// ---------------------------------------------------------------------------
// Kernel P: precompute the exp2 coefficients once (keeps the score kernel free
// of IEEE div chains; the values land in SGPRs there via scalar loads).
// ---------------------------------------------------------------------------
__global__ void knrm_prep(const float* __restrict__ sigmas, float* __restrict__ c2) {
    int i = threadIdx.x;
    if (i < NKERN) {
        float s = sigmas[i];
        c2[i] = -0.72134752f / (s * s);   // -0.5*log2(e)/sigma^2
    }
}

// ---------------------------------------------------------------------------
// Kernel A: L2-normalize embedding rows, write f16 padded table into workspace.
// One wave (32 lanes) per row, 8 rows per 256-thread block.
// ---------------------------------------------------------------------------
__global__ __launch_bounds__(256) void knrm_normalize(const float* __restrict__ emb,
                                                      _Float16* __restrict__ embn) {
    const int wave = threadIdx.x >> 5;
    const int lane = threadIdx.x & 31;
    const int row  = blockIdx.x * 8 + wave;
    if (row >= VOCAB) return;

    const float* src = emb + (size_t)row * D_IN;
    float ss = 0.0f;
    for (int j = lane; j < D_IN; j += 32) {
        float v = src[j];
        ss += v * v;
    }
#pragma unroll
    for (int m = 16; m >= 1; m >>= 1) ss += __shfl_xor(ss, m, 32);

    // v_rcp_f32: ~1ulp, scale error ~1e-7 -> sim error negligible
    const float scale = __builtin_amdgcn_rcpf(sqrtf(ss) + 1e-9f);
    _Float16* dst = embn + (size_t)row * D_PAD;
    for (int j = lane; j < D_PAD; j += 32) {
        float v = (j < D_IN) ? src[j] * scale : 0.0f;
        dst[j] = (_Float16)v;
    }
}

// ---------------------------------------------------------------------------
// Kernel B: one workgroup per (batch, side).
// M=16 (query), N=1024 (doc), K=320 GEMM via v_wmma_f32_16x16x32_f16 with a
// software-pipelined B-gather (register WAR rotation), A re-read from LDS per
// tile to keep VGPR pressure < 256, then RBF bank + masked log head.
// ---------------------------------------------------------------------------
__global__ __launch_bounds__(256) void knrm_score(
    const int* __restrict__ posdoc, const int* __restrict__ negdoc,
    const int* __restrict__ query,  const _Float16* __restrict__ embn,
    const float* __restrict__ mus,  const float* __restrict__ c2tab,
    const float* __restrict__ W,    const float* __restrict__ bias,
    float* __restrict__ out) {

    __shared__ __align__(32) _Float16 sQ[Q_LEN * D_PAD]; // 10 KB staged A operand
    __shared__ float sDocK[Q_LEN * NKERN];
    __shared__ float sSimSum[Q_LEN];
    __shared__ int   sQtok[Q_LEN];

    const int batch = blockIdx.x;
    const int side  = blockIdx.y;                 // 0 = posdoc, 1 = negdoc
    const int* doc  = side ? negdoc : posdoc;
    const int tid   = threadIdx.x;
    const int lane  = tid & 31;
    const int wave  = tid >> 5;

    if (tid < Q_LEN) {
        sQtok[tid]   = query[batch * Q_LEN + tid];
        sSimSum[tid] = 0.0f;
    }
    if (tid < Q_LEN * NKERN) sDocK[tid] = 0.0f;
    __syncthreads();

    // stage the 16 normalized query rows (f16, already padded) into LDS
    for (int i = tid; i < Q_LEN * D_PAD; i += 256) {
        int r = i / D_PAD, c = i - r * D_PAD;
        sQ[i] = embn[(size_t)sQtok[r] * D_PAD + c];
    }
    __syncthreads();

    const int  nlocal = lane & 15;
    const bool hi     = lane >= 16;
    const int  kofsB  = hi ? 16 : 0;              // B frag K base (halves)

    // uniform RBF parameters -> scalar loads -> SGPRs (no VALU, no VGPRs)
    float mu[NKERN], C2[NKERN];
#pragma unroll
    for (int kk = 0; kk < NKERN; ++kk) {
        mu[kk] = mus[kk];
        C2[kk] = c2tab[kk];
    }

    // query-row pad mask (m = r + 8*hi per C-fragment layout)
    float qvalid[8];
#pragma unroll
    for (int r = 0; r < 8; ++r)
        qvalid[r] = (sQtok[r + (hi ? 8 : 0)] != 0) ? 1.0f : 0.0f;

    float acc[8][NKERN];
    float ssum[8];
#pragma unroll
    for (int r = 0; r < 8; ++r) {
        ssum[r] = 0.0f;
#pragma unroll
        for (int kk = 0; kk < NKERN; ++kk) acc[r][kk] = 0.0f;
    }

    // A fragment LDS base (contiguous k order, same as B: lane m=lane&15,
    // low lanes K=0..15, high lanes K=16..31)
    const unsigned abase = (unsigned)(nlocal * D_PAD + (hi ? 16 : 0));

    // ---- all 8 doc tokens for this wave up front (kills token->row dep chain)
    const int* drow = doc + batch * T_LEN + wave * 128 + nlocal;
    int dtoks[8];
#pragma unroll
    for (int t = 0; t < 8; ++t) dtoks[t] = drow[t * 16];

    // ---- prologue: B fragments of tile 0
    v16h bcur[10];
    {
        const _Float16* p0 = embn + (size_t)dtoks[0] * D_PAD + kofsB;
#pragma unroll
        for (int ks = 0; ks < 10; ++ks)
            bcur[ks] = *(const v16h*)(p0 + ks * 32);   // 2x global_load_b128 each
    }

    // ---- pipelined tile loop: wmma on bcur[ks], then immediately refill
    // bcur[ks] with tile t+1's fragment (register WAR keeps the pipeline).
#pragma unroll
    for (int t = 0; t < 8; ++t) {
        // launder the LDS base per tile so the A reads are NOT CSE'd/hoisted
        // into 80 live VGPRs across all tiles (keeps pressure < 256).
        unsigned ab = abase;
        asm("" : "+v"(ab));
        const _Float16* arow = sQ + ab;

        v8f c = {};
#pragma unroll
        for (int ks = 0; ks < 10; ++ks) {
            const _Float16* ap = arow + ks * 32;
            v8h alo = *(const v8h*)ap;                // ds_load_b128
            v8h ahv = *(const v8h*)(ap + 8);          // ds_load_b128
            v16h a;
#pragma unroll
            for (int i = 0; i < 8; ++i) { a[i] = alo[i]; a[i + 8] = ahv[i]; }

            c = __builtin_amdgcn_wmma_f32_16x16x32_f16(
                    false, a, false, bcur[ks], (short)0, c, false, false);
            if (t < 7) {   // prefetch next tile's fragment into the just-freed regs
                const _Float16* pn = embn + (size_t)dtoks[t + 1] * D_PAD + kofsB;
                bcur[ks] = *(const v16h*)(pn + ks * 32);
            }
        }

        // epilogue: pad-mask (masked pairs stay sim=0 but still feed kernels);
        // its ~88 v_exp+FMA per lane also hides the tile t+1 gather latency.
        const float dm = (dtoks[t] != 0) ? 1.0f : 0.0f;
#pragma unroll
        for (int r = 0; r < 8; ++r) {
            float sim = c[r] * (dm * qvalid[r]);
            ssum[r] += sim;
#pragma unroll
            for (int kk = 0; kk < NKERN; ++kk) {
                float df = sim - mu[kk];
                acc[r][kk] += __builtin_amdgcn_exp2f(df * df * C2[kk]); // v_exp_f32
            }
        }
    }

    // reduce across the 16 N-lanes of each half-wave (xor masks < 16 stay in-half)
#pragma unroll
    for (int r = 0; r < 8; ++r) {
#pragma unroll
        for (int m = 8; m >= 1; m >>= 1) ssum[r] += __shfl_xor(ssum[r], m, 32);
#pragma unroll
        for (int kk = 0; kk < NKERN; ++kk) {
#pragma unroll
            for (int m = 8; m >= 1; m >>= 1) acc[r][kk] += __shfl_xor(acc[r][kk], m, 32);
        }
    }

    // commit per-wave partials (lane 0 -> rows 0..7, lane 16 -> rows 8..15)
    if (nlocal == 0) {
        const int mbase = hi ? 8 : 0;
#pragma unroll
        for (int r = 0; r < 8; ++r) {
            atomicAdd(&sSimSum[mbase + r], ssum[r]);                  // ds_add_f32
#pragma unroll
            for (int kk = 0; kk < NKERN; ++kk)
                atomicAdd(&sDocK[(mbase + r) * NKERN + kk], acc[r][kk]);
        }
    }
    __syncthreads();

    // scoring head: lane kk<11 sums masked logs over the 16 query rows
    if (wave == 0) {
        float qk = 0.0f;
        if (lane < NKERN) {
            const float w = W[lane];
#pragma unroll 1
            for (int m = 0; m < Q_LEN; ++m) {
                if (sSimSum[m] != 0.0f)
                    qk += __logf(sDocK[m * NKERN + lane] + 1e-6f);
            }
            qk *= w;
        }
#pragma unroll
        for (int m = 8; m >= 1; m >>= 1) qk += __shfl_xor(qk, m, 32);
        if (lane == 0) out[batch * 2 + side] = qk + bias[0];
    }
}

// ---------------------------------------------------------------------------
extern "C" void kernel_launch(void* const* d_in, const int* in_sizes, int n_in,
                              void* d_out, int out_size, void* d_ws, size_t ws_size,
                              hipStream_t stream) {
    const int*   posdoc = (const int*)d_in[0];
    const int*   negdoc = (const int*)d_in[1];
    const int*   query  = (const int*)d_in[2];
    /* d_in[3] = query_idf: unused by the reference computation */
    const float* emb    = (const float*)d_in[4];
    const float* mus    = (const float*)d_in[5];
    const float* sig    = (const float*)d_in[6];
    const float* W      = (const float*)d_in[7];
    const float* bb     = (const float*)d_in[8];
    float*       out    = (float*)d_out;

    _Float16* embn = (_Float16*)d_ws;                       // 32 MB f16 table
    float*    c2   = (float*)((char*)d_ws + EMBN_BYTES);    // 11 coefficients

    knrm_prep<<<1, 16, 0, stream>>>(sig, c2);
    knrm_normalize<<<VOCAB / 8, 256, 0, stream>>>(emb, embn);

    dim3 grid(BATCH, 2);
    knrm_score<<<grid, 256, 0, stream>>>(posdoc, negdoc, query, embn,
                                         mus, c2, W, bb, out);
}
